// EC_SparseMoeBlock_29729763623449
// MI455X (gfx1250) — compile-verified, matrix operand order checked by reading, not run on hardware
//
#include <hip/hip_runtime.h>
#include <hip/hip_bf16.h>
#include <math.h>

// ---------------- problem constants ----------------
#define BB     4
#define SS     4096
#define DD     1024
#define EE     16
#define FF     2048      // F_EXP == F_SH
#define CAP    512       // expert capacity = S*2/E

typedef __attribute__((ext_vector_type(16))) _Float16 v16h;
typedef __attribute__((ext_vector_type(8)))  _Float16 v8h;
typedef __attribute__((ext_vector_type(8)))  float    v8f;

// ---------------- WMMA helpers ----------------
__device__ __forceinline__ v8f wmma16(v16h a, v16h b, v8f c) {
  // D = A(16x32 f16) x B(32x16 f16) + C(16x16 f32)
  return __builtin_amdgcn_wmma_f32_16x16x32_f16(false, a, false, b,
                                                (short)0, c, false, false);
}

// A fragment: lane (m = lane&15) holds K = kb..kb+7 and 16+kb..16+kb+7,
// kb = (lane>>4)*8. Tile stored row-major [16][STRIDE] halves (K contiguous).
template<int STRIDE>
__device__ __forceinline__ v16h load_a_frag(const _Float16* tile, int lane, int kbase) {
  const int m  = lane & 15;
  const int kb = (lane >> 4) << 3;
  const v8h lo = *(const v8h*)(tile + m * STRIDE + kbase + kb);
  const v8h hi = *(const v8h*)(tile + m * STRIDE + kbase + 16 + kb);
  v16h r;
#pragma unroll
  for (int i = 0; i < 8; ++i) { r[i] = lo[i]; r[8 + i] = hi[i]; }
  return r;
}

// B fragment: lane (n = ntile*16 + (lane&15)) holds same K pattern.
// Tile stored [n][k] (k contiguous), i.e. transposed K x N.
template<int STRIDE>
__device__ __forceinline__ v16h load_b_frag(const _Float16* tile, int lane, int ntile) {
  const int n  = ntile * 16 + (lane & 15);
  const int kb = (lane >> 4) << 3;
  const v8h lo = *(const v8h*)(tile + n * STRIDE + kb);
  const v8h hi = *(const v8h*)(tile + n * STRIDE + 16 + kb);
  v16h r;
#pragma unroll
  for (int i = 0; i < 8; ++i) { r[i] = lo[i]; r[8 + i] = hi[i]; }
  return r;
}

__device__ __forceinline__ float gelu_exact(float x) {
  return 0.5f * x * (1.0f + erff(x * 0.70710678118654752f));
}

// ---------------- kernel 1: gate logits + softmax -> scores (B,E,S) ----------------
// 8 waves/block, each wave handles 2 tokens; lane e = lane&15 owns one expert dot.
__global__ __launch_bounds__(256) void gate_softmax_kernel(
    const float* __restrict__ x, const float* __restrict__ gw,
    float* __restrict__ scores) {
  const int tid  = threadIdx.x;
  const int w    = tid >> 5;
  const int lane = tid & 31;
  const int ft   = blockIdx.x * 16 + w * 2 + (lane >> 4);   // flat token
  const int e    = lane & 15;

  const float* xr = x  + (size_t)ft * DD;
  const float* wr = gw + (size_t)e  * DD;
  float acc = 0.f;
#pragma unroll 4
  for (int d = 0; d < DD; d += 4) {
    float4 xv = *(const float4*)(xr + d);
    float4 wv = *(const float4*)(wr + d);
    acc = fmaf(xv.x, wv.x, acc);
    acc = fmaf(xv.y, wv.y, acc);
    acc = fmaf(xv.z, wv.z, acc);
    acc = fmaf(xv.w, wv.w, acc);
  }
  // softmax across the 16-lane expert group (xor 1,2,4,8 stays in-group)
  float m = acc;
#pragma unroll
  for (int off = 1; off <= 8; off <<= 1) m = fmaxf(m, __shfl_xor(m, off, 32));
  float p = expf(acc - m);
  float s = p;
#pragma unroll
  for (int off = 1; off <= 8; off <<= 1) s += __shfl_xor(s, off, 32);
  const int b  = ft >> 12;           // / SS
  const int si = ft & (SS - 1);
  scores[((size_t)b * EE + e) * SS + si] = p / s;
}

// ---------------- kernel 2: per-(b,e) top-512 via LDS bitonic sort ----------------
// key = (orderable(score) << 12) | (4095 - idx): ascending sort; ties -> lower idx wins.
__global__ __launch_bounds__(512) void topk_kernel(
    const float* __restrict__ scores,
    int* __restrict__ topkIdx, float* __restrict__ topkW) {
  __shared__ unsigned long long keys[SS];
  const int be  = blockIdx.x;
  const int tid = threadIdx.x;

  for (int i = tid; i < SS; i += 512) {
    const float f = scores[(size_t)be * SS + i];
    unsigned u = __float_as_uint(f);
    unsigned fk = (u & 0x80000000u) ? ~u : (u | 0x80000000u);
    keys[i] = (((unsigned long long)fk) << 12) | (unsigned long long)(SS - 1 - i);
  }
  __syncthreads();

  for (int k = 2; k <= SS; k <<= 1) {
    for (int j = k >> 1; j > 0; j >>= 1) {
      for (int i = tid; i < SS; i += 512) {
        const int ixj = i ^ j;
        if (ixj > i) {
          const bool up = ((i & k) == 0);
          unsigned long long a = keys[i], b = keys[ixj];
          if ((a > b) == up) { keys[i] = b; keys[ixj] = a; }
        }
      }
      __syncthreads();
    }
  }

  // top-CAP in descending order
  {
    const int c = tid;  // 512 threads == CAP
    const unsigned long long key = keys[SS - 1 - c];
    const int idx = (SS - 1) - (int)(key & 0xFFFull);
    unsigned fk = (unsigned)(key >> 12);
    unsigned u  = (fk & 0x80000000u) ? (fk & 0x7FFFFFFFu) : ~fk;
    topkIdx[(size_t)be * CAP + c] = idx;
    topkW [(size_t)be * CAP + c] = __uint_as_float(u);
  }
}

// ---------------- FFN core (shared by kernels 3 & 4) ----------------
// Block = 256 threads = 8 waves. 16 token rows per block, full F loop in 8
// chunks of 256; down-proj accumulated in 8 persistent v8f tiles per wave.
// LDS: A 16x(32+8), Bg/Bu 256x(32+8), H 16x(256+8)  ~50KB.

struct FfnSmem {
  __align__(16) _Float16 sA[16 * 40];
  __align__(16) _Float16 sBg[256 * 40];
  __align__(16) _Float16 sBu[256 * 40];
  __align__(16) _Float16 sH[16 * 264];
};

// rowGlob[r]: flat row index into x (B*S rows). Returns accO[8] per wave.
__device__ __forceinline__ void ffn_core(
    FfnSmem& sm, const float* __restrict__ x, const int* __restrict__ rowGlob,
    const float* __restrict__ Wgp, const float* __restrict__ Wup,
    const float* __restrict__ Wdp, int tid, v8f accO[8]) {
  const int w    = tid >> 5;
  const int lane = tid & 31;

#pragma unroll
  for (int i = 0; i < 8; ++i)
#pragma unroll
    for (int j = 0; j < 8; ++j) accO[i][j] = 0.f;

  for (int fc = 0; fc < 8; ++fc) {
    const int f0 = fc * 256;
    v8f accG[2], accU[2];
#pragma unroll
    for (int t = 0; t < 2; ++t)
#pragma unroll
      for (int j = 0; j < 8; ++j) { accG[t][j] = 0.f; accU[t][j] = 0.f; }

    // ---- phase 1: g,u over K = D ----
    for (int kk = 0; kk < DD; kk += 32) {
      {   // stage A: 16 rows x 32 k (fp32 -> f16), 2 elems/thread
        const int r  = tid >> 4;
        const int c2 = (tid & 15) * 2;
        const float2 v = *(const float2*)(x + (size_t)rowGlob[r] * DD + kk + c2);
        sm.sA[r * 40 + c2]     = (_Float16)v.x;
        sm.sA[r * 40 + c2 + 1] = (_Float16)v.y;
      }
      {   // stage Bg/Bu: thread t owns weight row f0+t, 32 k values
        const float* gsrc = Wgp + (size_t)(f0 + tid) * DD + kk;
        const float* usrc = Wup + (size_t)(f0 + tid) * DD + kk;
        if (kk + 32 < DD) {
          __builtin_prefetch(gsrc + 32, 0, 1);
          __builtin_prefetch(usrc + 32, 0, 1);
        }
#pragma unroll
        for (int q = 0; q < 8; ++q) {
          const float4 gv = *(const float4*)(gsrc + q * 4);
          const float4 uv = *(const float4*)(usrc + q * 4);
          const int cb = q * 4;
          sm.sBg[tid * 40 + cb + 0] = (_Float16)gv.x;
          sm.sBg[tid * 40 + cb + 1] = (_Float16)gv.y;
          sm.sBg[tid * 40 + cb + 2] = (_Float16)gv.z;
          sm.sBg[tid * 40 + cb + 3] = (_Float16)gv.w;
          sm.sBu[tid * 40 + cb + 0] = (_Float16)uv.x;
          sm.sBu[tid * 40 + cb + 1] = (_Float16)uv.y;
          sm.sBu[tid * 40 + cb + 2] = (_Float16)uv.z;
          sm.sBu[tid * 40 + cb + 3] = (_Float16)uv.w;
        }
      }
      __syncthreads();
      const v16h aF = load_a_frag<40>(sm.sA, lane, 0);
#pragma unroll
      for (int t = 0; t < 2; ++t) {
        const v16h bG = load_b_frag<40>(sm.sBg, lane, w * 2 + t);
        accG[t] = wmma16(aF, bG, accG[t]);
        const v16h bU = load_b_frag<40>(sm.sBu, lane, w * 2 + t);
        accU[t] = wmma16(aF, bU, accU[t]);
      }
      __syncthreads();
    }

    // ---- activation: h = gelu(g)*u  -> sH (f16) ----
#pragma unroll
    for (int t = 0; t < 2; ++t) {
      const int col = (w * 2 + t) * 16 + (lane & 15);
#pragma unroll
      for (int i = 0; i < 8; ++i) {
        const int row = ((lane >> 4) << 3) + i;
        sm.sH[row * 264 + col] = (_Float16)(gelu_exact(accG[t][i]) * accU[t][i]);
      }
    }
    __syncthreads();

    // ---- phase 2: accO += h_chunk @ Wd[:, f-chunk]^T ----
    for (int nc = 0; nc < 4; ++nc) {
      for (int k2 = 0; k2 < 256; k2 += 32) {
        const float* dsrc = Wdp + (size_t)(nc * 256 + tid) * FF + f0 + k2;
        if (k2 + 32 < 256) __builtin_prefetch(dsrc + 32, 0, 1);
#pragma unroll
        for (int q = 0; q < 8; ++q) {
          const float4 dv = *(const float4*)(dsrc + q * 4);
          const int cb = q * 4;
          sm.sBg[tid * 40 + cb + 0] = (_Float16)dv.x;
          sm.sBg[tid * 40 + cb + 1] = (_Float16)dv.y;
          sm.sBg[tid * 40 + cb + 2] = (_Float16)dv.z;
          sm.sBg[tid * 40 + cb + 3] = (_Float16)dv.w;
        }
        __syncthreads();
        const v16h aF = load_a_frag<264>(sm.sH, lane, k2);
#pragma unroll
        for (int t = 0; t < 2; ++t) {
          const v16h bD = load_b_frag<40>(sm.sBg, lane, w * 2 + t);
          accO[nc * 2 + t] = wmma16(aF, bD, accO[nc * 2 + t]);
        }
        __syncthreads();
      }
    }
  }
}

// ---------------- kernel 3: shared expert (dense, writes out) ----------------
__global__ __launch_bounds__(256) void shared_ffn_kernel(
    const float* __restrict__ x, const float* __restrict__ Wsg,
    const float* __restrict__ Wsu, const float* __restrict__ Wsd,
    float* __restrict__ out) {
  __shared__ FfnSmem sm;
  __shared__ int rowGlob[16];
  const int tid = threadIdx.x;
  if (tid < 16) rowGlob[tid] = blockIdx.x * 16 + tid;
  __syncthreads();

  v8f accO[8];
  ffn_core(sm, x, rowGlob, Wsg, Wsu, Wsd, tid, accO);

  const int w = tid >> 5, lane = tid & 31;
#pragma unroll
  for (int nc = 0; nc < 4; ++nc)
#pragma unroll
    for (int t = 0; t < 2; ++t) {
      const int col = nc * 256 + (w * 2 + t) * 16 + (lane & 15);
#pragma unroll
      for (int i = 0; i < 8; ++i) {
        const int row = ((lane >> 4) << 3) + i;
        out[(size_t)rowGlob[row] * DD + col] = accO[nc * 2 + t][i];
      }
    }
}

// ---------------- kernel 4: expert FFN (gather, scale, atomic scatter-add) ----------------
__global__ __launch_bounds__(256) void expert_ffn_kernel(
    const float* __restrict__ x, const float* __restrict__ Wg,
    const float* __restrict__ Wu, const float* __restrict__ Wd,
    const int* __restrict__ topkIdx, const float* __restrict__ topkW,
    float* __restrict__ out) {
  __shared__ FfnSmem sm;
  __shared__ int   rowGlob[16];
  __shared__ float sW[16];
  const int be = blockIdx.y;          // b*16 + e
  const int b  = be >> 4;
  const int e  = be & 15;
  const int rc = blockIdx.x;          // row chunk 0..31
  const int tid = threadIdx.x;

  if (tid < 16) {
    const int c = rc * 16 + tid;
    rowGlob[tid] = b * SS + topkIdx[(size_t)be * CAP + c];
    sW[tid]      = topkW[(size_t)be * CAP + c];
  }
  __syncthreads();

  const float* Wgp = Wg + (size_t)e * FF * DD;
  const float* Wup = Wu + (size_t)e * FF * DD;
  const float* Wdp = Wd + (size_t)e * DD * FF;

  v8f accO[8];
  ffn_core(sm, x, rowGlob, Wgp, Wup, Wdp, tid, accO);

  const int w = tid >> 5, lane = tid & 31;
#pragma unroll
  for (int nc = 0; nc < 4; ++nc)
#pragma unroll
    for (int t = 0; t < 2; ++t) {
      const int col = nc * 256 + (w * 2 + t) * 16 + (lane & 15);
#pragma unroll
      for (int i = 0; i < 8; ++i) {
        const int row = ((lane >> 4) << 3) + i;
        atomicAdd(out + (size_t)rowGlob[row] * DD + col,
                  accO[nc * 2 + t][i] * sW[row]);
      }
    }
}

// ---------------- host launcher ----------------
extern "C" void kernel_launch(void* const* d_in, const int* in_sizes, int n_in,
                              void* d_out, int out_size, void* d_ws, size_t ws_size,
                              hipStream_t stream) {
  const float* x    = (const float*)d_in[0];   // (B,S,D)
  const float* gw   = (const float*)d_in[1];   // (E,D)
  const float* Wg   = (const float*)d_in[2];   // (E,F,D)
  const float* Wu   = (const float*)d_in[3];   // (E,F,D)
  const float* Wd   = (const float*)d_in[4];   // (E,D,F)
  const float* Wsg  = (const float*)d_in[5];   // (F,D)
  const float* Wsu  = (const float*)d_in[6];   // (F,D)
  const float* Wsd  = (const float*)d_in[7];   // (D,F)
  float* out = (float*)d_out;

  // workspace layout
  float* scores  = (float*)d_ws;                                   // B*E*S
  char*  p1      = (char*)d_ws + (size_t)BB * EE * SS * sizeof(float);
  int*   topkIdx = (int*)p1;                                       // B*E*CAP
  float* topkW   = (float*)(p1 + (size_t)BB * EE * CAP * sizeof(int));

  // 1) gate + softmax
  gate_softmax_kernel<<<(BB * SS) / 16, 256, 0, stream>>>(x, gw, scores);
  // 2) per-(b,e) top-512
  topk_kernel<<<BB * EE, 512, 0, stream>>>(scores, topkIdx, topkW);
  // 3) shared expert writes out (initializer for scatter-add)
  shared_ffn_kernel<<<(BB * SS) / 16, 256, 0, stream>>>(x, Wsg, Wsu, Wsd, out);
  // 4) expert FFN: 32 row-chunks x (B*E) blocks, atomic scatter-add
  expert_ffn_kernel<<<dim3(32, BB * EE), 256, 0, stream>>>(
      x, Wg, Wu, Wd, topkIdx, topkW, out);
}
